// TransformerBlock_67894843015264
// MI455X (gfx1250) — matrix operand order repaired and checked by reading, not compiled, not run on hardware
//
#include <hip/hip_runtime.h>
#include <math.h>

#define DIMV  768
#define HEADS 12
#define DHEAD 64
#define BB    8
#define NSEQ  1024
#define MROWS (BB * NSEQ) /* 8192 */

typedef __bf16 bf16_t;
typedef __attribute__((ext_vector_type(8)))  __bf16 v8bf;
typedef __attribute__((ext_vector_type(16))) __bf16 v16bf;
typedef __attribute__((ext_vector_type(8)))  float  v8f;

union BF16x16 { v16bf f; v8bf h[2]; };
union F8      { v8f v; float f[8]; };

__device__ __forceinline__ v8f zero8() { v8f z = {0.f,0.f,0.f,0.f,0.f,0.f,0.f,0.f}; return z; }

// ---------------------------------------------------------------------------
// CDNA5 async global->LDS copy (ASYNCcnt-tracked, no VGPR data round-trip).
// Per ISA 10.x: each lane copies 16 bytes from its global address to its LDS address.
// Generic LDS pointers truncate to the DS byte address (aperture: addr[31:0]).
__device__ __forceinline__ void async_load_b128(void* lds_dst, const void* gsrc) {
  unsigned l = (unsigned)(uintptr_t)lds_dst;
  unsigned long long g = (unsigned long long)(uintptr_t)gsrc;
  asm volatile("global_load_async_to_lds_b128 %0, %1, off"
               :: "v"(l), "v"(g) : "memory");
}
__device__ __forceinline__ void wait_async() {
  asm volatile("s_wait_asynccnt 0" ::: "memory");
}

// ---------------------------------------------------------------------------
// Fragment loaders for WMMA operands (wave32), contiguous k order used for both A and B.
// A (16-bit, 16x32, row-major src, stride ld):
//   lanes 0-15 : row=lane,    elems = K 0..15
//   lanes16-31 : row=lane-16, elems = K 16..31
__device__ __forceinline__ v16bf load_a_frag(const bf16_t* tile, int ld) {
  const int lane = threadIdx.x & 31;
  const int row  = lane & 15;
  const int k0   = (lane & 16) ? 16 : 0;
  const bf16_t* p = tile + (size_t)row * ld + k0;
  BF16x16 u;
  u.h[0] = *(const v8bf*)(p);
  u.h[1] = *(const v8bf*)(p + 8);
  return u.f;
}
// B (16-bit, 32x16). Source is B^T row-major [N rows, K], stride ld:
//   lanes 0-15 : col=lane,    elems = K 0..15
//   lanes16-31 : col=lane-16, elems = K 16..31
__device__ __forceinline__ v16bf load_b_frag(const bf16_t* tile, int ld) {
  const int lane = threadIdx.x & 31;
  const int row  = lane & 15;
  const int k0   = (lane & 16) ? 16 : 0;
  const bf16_t* p = tile + (size_t)row * ld + k0;
  BF16x16 u;
  u.h[0] = *(const v8bf*)(p);
  u.h[1] = *(const v8bf*)(p + 8);
  return u.f;
}

__device__ __forceinline__ v8f wmma_bf16(v16bf a, v16bf b, v8f c) {
  return __builtin_amdgcn_wmma_f32_16x16x32_bf16(false, a, false, b, (short)0, c,
                                                 false, false);
}

// ---------------------------------------------------------------------------
// Weight convert: f32 W[K,N] -> bf16 W^T[N,K]
__global__ __launch_bounds__(256) void cvt_wt_kernel(const float* __restrict__ W,
                                                     bf16_t* __restrict__ Wt,
                                                     int K, int N) {
  size_t idx = (size_t)blockIdx.x * 256 + threadIdx.x;
  if (idx >= (size_t)K * N) return;
  int k = (int)(idx / (size_t)N);
  int n = (int)(idx - (size_t)k * N);
  Wt[(size_t)n * K + k] = (bf16_t)W[idx];
}

// ---------------------------------------------------------------------------
// LayerNorm over 768, f32 in -> bf16 out. One block (8 waves) per row.
__global__ __launch_bounds__(256) void ln_bf16_kernel(const float* __restrict__ x,
                                                      const float* __restrict__ g,
                                                      const float* __restrict__ bta,
                                                      bf16_t* __restrict__ out) {
  __shared__ float ssum[256];
  __shared__ float ssq[256];
  const int row = blockIdx.x;
  const float* xr = x + (size_t)row * DIMV;
  float v[3], s = 0.f, q = 0.f;
  for (int i = 0; i < 3; ++i) {
    v[i] = xr[threadIdx.x + i * 256];
    s += v[i]; q += v[i] * v[i];
  }
  ssum[threadIdx.x] = s; ssq[threadIdx.x] = q;
  __syncthreads();
  for (int off = 128; off > 0; off >>= 1) {
    if ((int)threadIdx.x < off) {
      ssum[threadIdx.x] += ssum[threadIdx.x + off];
      ssq[threadIdx.x]  += ssq[threadIdx.x + off];
    }
    __syncthreads();
  }
  const float mu  = ssum[0] * (1.f / DIMV);
  const float var = ssq[0] * (1.f / DIMV) - mu * mu;
  const float rs  = rsqrtf(var + 1e-5f);
  for (int i = 0; i < 3; ++i) {
    int c = threadIdx.x + i * 256;
    out[(size_t)row * DIMV + c] = (bf16_t)((v[i] - mu) * rs * g[c] + bta[c]);
  }
}

// ---------------------------------------------------------------------------
// GEMM main loop: 128x64 block tile, 8 waves (4x2), 32x32 per wave.
// Double-buffered LDS filled by async global->LDS; one barrier per k-step.
// As: 2 stages of 128x32 bf16, Bs: 2 stages of 64x32 bf16.
__device__ __forceinline__ void gemm_mainloop(const bf16_t* __restrict__ A, int lda,
                                              const bf16_t* __restrict__ Bt, int ldb,
                                              int K, int bm, int bn,
                                              bf16_t* As, bf16_t* Bs,
                                              v8f acc[2][2]) {
  const int tid  = threadIdx.x;
  const int wave = tid >> 5;
  const int rw   = wave >> 1;        // 0..3 : 32-row slab
  const int cw   = wave & 1;         // 0..1 : 32-col slab
  // A staging: 128x32 = 256 x (16 bf16); thread -> row=tid>>1, half=(tid&1)*16
  const int ar = tid >> 1;
  const int ac = (tid & 1) * 16;
  const bf16_t* agp = A + (size_t)(bm + ar) * lda + ac;
  // B staging: 64x32 = 256 x (8 bf16); thread -> row=tid>>2, seg=(tid&3)*8
  const int br = tid >> 2;
  const int bc = (tid & 3) * 8;
  const bf16_t* bgp = Bt + (size_t)(bn + br) * ldb + bc;

  // prologue: fill stage 0
  async_load_b128(As + ar * 32 + ac,     agp);
  async_load_b128(As + ar * 32 + ac + 8, agp + 8);
  async_load_b128(Bs + br * 32 + bc,     bgp);
  wait_async();
  __syncthreads();

  int st = 0;
  for (int k0 = 0; k0 < K; k0 += 32, st ^= 1) {
    if (k0 + 32 < K) {   // issue next stage while computing this one
      bf16_t* alp = As + (st ^ 1) * (128 * 32) + ar * 32 + ac;
      bf16_t* blp = Bs + (st ^ 1) * (64 * 32)  + br * 32 + bc;
      async_load_b128(alp,     agp + k0 + 32);
      async_load_b128(alp + 8, agp + k0 + 32 + 8);
      async_load_b128(blp,     bgp + k0 + 32);
    }
    const bf16_t* Ab = As + st * (128 * 32);
    const bf16_t* Bb = Bs + st * (64 * 32);
    v16bf a0 = load_a_frag(Ab + (rw * 32) * 32, 32);
    v16bf a1 = load_a_frag(Ab + (rw * 32 + 16) * 32, 32);
    v16bf b0 = load_b_frag(Bb + (cw * 32) * 32, 32);
    v16bf b1 = load_b_frag(Bb + (cw * 32 + 16) * 32, 32);
    acc[0][0] = wmma_bf16(a0, b0, acc[0][0]);
    acc[0][1] = wmma_bf16(a0, b1, acc[0][1]);
    acc[1][0] = wmma_bf16(a1, b0, acc[1][0]);
    acc[1][1] = wmma_bf16(a1, b1, acc[1][1]);
    wait_async();        // this wave's next-stage fills are in LDS
    __syncthreads();     // everyone done reading stage st / filling st^1
  }
}

// ---------------------------------------------------------------------------
// QKV GEMM: hbf[8192,768] @ WqkvT -> scatter to Q/K [b,h,n,d] and V^T [b,h,d,n]
__global__ __launch_bounds__(256) void gemm_qkv_kernel(const bf16_t* __restrict__ A,
                                                       const bf16_t* __restrict__ Bt,
                                                       bf16_t* __restrict__ Qo,
                                                       bf16_t* __restrict__ Ko,
                                                       bf16_t* __restrict__ Vt) {
  __shared__ bf16_t As[2 * 128 * 32];
  __shared__ bf16_t Bs[2 * 64 * 32];
  const int bm = blockIdx.x * 128, bn = blockIdx.y * 64;
  v8f acc[2][2] = {{zero8(), zero8()}, {zero8(), zero8()}};
  gemm_mainloop(A, DIMV, Bt, DIMV, DIMV, bm, bn, As, Bs, acc);
  const int lane = threadIdx.x & 31;
  const int wave = threadIdx.x >> 5;
  const int rw = wave >> 1, cw = wave & 1;
  const int colb = lane & 15;
  const int rowoff = (lane & 16) ? 8 : 0;
  F8 u[2][2];
  u[0][0].v = acc[0][0]; u[0][1].v = acc[0][1];
  u[1][0].v = acc[1][0]; u[1][1].v = acc[1][1];
  for (int ai = 0; ai < 2; ++ai) {
    for (int j = 0; j < 8; ++j) {
      const int m = bm + rw * 32 + ai * 16 + rowoff + j;
      const int b = m >> 10, n = m & 1023;
      for (int bi = 0; bi < 2; ++bi) {
        const int col   = bn + cw * 32 + bi * 16 + colb;  // 0..2303
        const int which = col / DIMV;
        const int rem   = col - which * DIMV;
        const int hh = rem >> 6, d = rem & 63;
        const size_t bh = (size_t)b * HEADS + hh;
        const bf16_t val = (bf16_t)u[ai][bi].f[j];
        if (which == 0)       Qo[(bh * NSEQ + n) * DHEAD + d] = val;
        else if (which == 1)  Ko[(bh * NSEQ + n) * DHEAD + d] = val;
        else                  Vt[(bh * DHEAD + d) * NSEQ + n] = val;
      }
    }
  }
}

// ---------------------------------------------------------------------------
// Generic GEMM: C = A[8192,K]bf16 @ Bt[N,K]bf16, fused bias/GELU/residual,
// f32 and/or bf16 outputs.
__global__ __launch_bounds__(256) void gemm_std_kernel(const bf16_t* __restrict__ A, int lda,
                                                       const bf16_t* __restrict__ Bt, int ldb,
                                                       int N, int K,
                                                       const float* __restrict__ bias,
                                                       const float* __restrict__ residual,
                                                       int do_gelu,
                                                       float* __restrict__ outF,
                                                       bf16_t* __restrict__ outB) {
  __shared__ bf16_t As[2 * 128 * 32];
  __shared__ bf16_t Bs[2 * 64 * 32];
  const int bm = blockIdx.x * 128, bn = blockIdx.y * 64;
  v8f acc[2][2] = {{zero8(), zero8()}, {zero8(), zero8()}};
  gemm_mainloop(A, lda, Bt, ldb, K, bm, bn, As, Bs, acc);
  const int lane = threadIdx.x & 31;
  const int wave = threadIdx.x >> 5;
  const int rw = wave >> 1, cw = wave & 1;
  const int colb = lane & 15;
  const int rowoff = (lane & 16) ? 8 : 0;
  F8 u[2][2];
  u[0][0].v = acc[0][0]; u[0][1].v = acc[0][1];
  u[1][0].v = acc[1][0]; u[1][1].v = acc[1][1];
  for (int ai = 0; ai < 2; ++ai) {
    for (int j = 0; j < 8; ++j) {
      const int m = bm + rw * 32 + ai * 16 + rowoff + j;
      for (int bi = 0; bi < 2; ++bi) {
        const int col = bn + cw * 32 + bi * 16 + colb;
        float v = u[ai][bi].f[j];
        if (bias)     v += bias[col];
        if (do_gelu)  v = 0.5f * v * (1.0f + erff(v * 0.70710678118654752f));
        if (residual) v += residual[(size_t)m * N + col];
        if (outF) outF[(size_t)m * N + col] = v;
        if (outB) outB[(size_t)m * N + col] = (bf16_t)v;
      }
    }
  }
}

// ---------------------------------------------------------------------------
// Flash attention: one wave per 16-query tile per (b,h). Online softmax in f32.
__global__ __launch_bounds__(32) void attn_kernel(const bf16_t* __restrict__ Q,
                                                  const bf16_t* __restrict__ Kb,
                                                  const bf16_t* __restrict__ Vt,
                                                  bf16_t* __restrict__ O) {
  __shared__ bf16_t Ps[16 * 32];
  const int bh   = blockIdx.y;
  const int bidx = bh / HEADS, hh = bh - bidx * HEADS;
  const int q0   = blockIdx.x * 16;
  const int lane = threadIdx.x & 31;
  const bf16_t* qbase = Q  + ((size_t)bh * NSEQ + q0) * DHEAD;
  const bf16_t* kbase = Kb + (size_t)bh * NSEQ * DHEAD;
  const bf16_t* vbase = Vt + (size_t)bh * DHEAD * NSEQ;

  const v16bf aq0 = load_a_frag(qbase, DHEAD);        // d 0..31
  const v16bf aq1 = load_a_frag(qbase + 32, DHEAD);   // d 32..63

  float mrow[8], lrow[8];
  F8 o[4];
  for (int j = 0; j < 8; ++j) { mrow[j] = -3.0e38f; lrow[j] = 0.f; }
  for (int t = 0; t < 4; ++t) o[t].v = zero8();

  const int colb   = lane & 15;
  const int rowoff = (lane & 16) ? 8 : 0;

  for (int kc = 0; kc < NSEQ; kc += 32) {
    F8 s0, s1;
    {
      const bf16_t* kp0 = kbase + (size_t)kc * DHEAD;
      v8f z = zero8();
      z = wmma_bf16(aq0, load_b_frag(kp0, DHEAD), z);
      z = wmma_bf16(aq1, load_b_frag(kp0 + 32, DHEAD), z);
      s0.v = z;
      const bf16_t* kp1 = kp0 + 16 * DHEAD;
      v8f y = zero8();
      y = wmma_bf16(aq0, load_b_frag(kp1, DHEAD), y);
      y = wmma_bf16(aq1, load_b_frag(kp1 + 32, DHEAD), y);
      s1.v = y;
    }
    for (int j = 0; j < 8; ++j) {
      float a0 = s0.f[j] * 0.125f;    // DIM_HEAD^-0.5
      float a1 = s1.f[j] * 0.125f;
      float mx = fmaxf(a0, a1);
      // row (fixed j) spans one 16-lane half: reduce with xor masks 8..1
      for (int msk = 8; msk >= 1; msk >>= 1) mx = fmaxf(mx, __shfl_xor(mx, msk, 32));
      float mn = fmaxf(mrow[j], mx);
      float alpha = __expf(mrow[j] - mn);
      mrow[j] = mn;
      float p0 = __expf(a0 - mn);
      float p1 = __expf(a1 - mn);
      float rs = p0 + p1;
      for (int msk = 8; msk >= 1; msk >>= 1) rs += __shfl_xor(rs, msk, 32);
      lrow[j] = lrow[j] * alpha + rs;
      for (int t = 0; t < 4; ++t) o[t].f[j] *= alpha;
      const int prow = rowoff + j;                 // C-layout row
      Ps[prow * 32 + colb]      = (bf16_t)p0;
      Ps[prow * 32 + 16 + colb] = (bf16_t)p1;
    }
    asm volatile("s_wait_dscnt 0" ::: "memory");   // P in LDS visible to loads
    const v16bf pa = load_a_frag(Ps, 32);          // re-layout C->A via LDS
    for (int t = 0; t < 4; ++t) {
      const bf16_t* vp = vbase + (size_t)(t * 16) * NSEQ + kc;
      o[t].v = wmma_bf16(pa, load_b_frag(vp, NSEQ), o[t].v);
    }
    asm volatile("s_wait_dscnt 0" ::: "memory");   // loads done before rewrite
  }
  for (int j = 0; j < 8; ++j) {
    const int n = q0 + rowoff + j;
    const float inv = 1.0f / lrow[j];
    bf16_t* orow = O + ((size_t)bidx * NSEQ + n) * DIMV + hh * DHEAD;
    for (int t = 0; t < 4; ++t)
      orow[t * 16 + colb] = (bf16_t)(o[t].f[j] * inv);
  }
}

// ---------------------------------------------------------------------------
extern "C" void kernel_launch(void* const* d_in, const int* in_sizes, int n_in,
                              void* d_out, int out_size, void* d_ws, size_t ws_size,
                              hipStream_t stream) {
  (void)in_sizes; (void)n_in; (void)out_size; (void)ws_size;
  const float* x     = (const float*)d_in[0];
  const float* ln1_g = (const float*)d_in[1];
  const float* ln1_b = (const float*)d_in[2];
  const float* ln2_g = (const float*)d_in[3];
  const float* ln2_b = (const float*)d_in[4];
  const float* w_qkv = (const float*)d_in[5];
  const float* w_out = (const float*)d_in[6];
  const float* b_out = (const float*)d_in[7];
  const float* w_ff1 = (const float*)d_in[8];
  const float* b_ff1 = (const float*)d_in[9];
  const float* w_ff2 = (const float*)d_in[10];
  const float* b_ff2 = (const float*)d_in[11];

  size_t off = 0;
  char* base = (char*)d_ws;
  auto alloc = [&](size_t bytes) -> char* {
    char* p = base + off;
    off += (bytes + 255) & ~(size_t)255;
    return p;
  };
  bf16_t* wqkv_t = (bf16_t*)alloc((size_t)2304 * 768 * 2);
  bf16_t* wout_t = (bf16_t*)alloc((size_t)768 * 768 * 2);
  bf16_t* wff1_t = (bf16_t*)alloc((size_t)3072 * 768 * 2);
  bf16_t* wff2_t = (bf16_t*)alloc((size_t)768 * 3072 * 2);
  bf16_t* hbf    = (bf16_t*)alloc((size_t)MROWS * DIMV * 2);
  bf16_t* qb     = (bf16_t*)alloc((size_t)MROWS * DIMV * 2);
  bf16_t* kb     = (bf16_t*)alloc((size_t)MROWS * DIMV * 2);
  bf16_t* vtb    = (bf16_t*)alloc((size_t)MROWS * DIMV * 2);
  bf16_t* attnb  = (bf16_t*)alloc((size_t)MROWS * DIMV * 2);
  float*  x2     = (float*) alloc((size_t)MROWS * DIMV * 4);
  bf16_t* ff1b   = (bf16_t*)alloc((size_t)MROWS * 3072 * 2);

  auto cdiv = [](size_t a, size_t b) { return (unsigned)((a + b - 1) / b); };

  // weight downconvert + transpose
  cvt_wt_kernel<<<cdiv((size_t)768 * 2304, 256), 256, 0, stream>>>(w_qkv, wqkv_t, 768, 2304);
  cvt_wt_kernel<<<cdiv((size_t)768 * 768,  256), 256, 0, stream>>>(w_out, wout_t, 768, 768);
  cvt_wt_kernel<<<cdiv((size_t)768 * 3072, 256), 256, 0, stream>>>(w_ff1, wff1_t, 768, 3072);
  cvt_wt_kernel<<<cdiv((size_t)3072 * 768, 256), 256, 0, stream>>>(w_ff2, wff2_t, 3072, 768);

  // h1 = LN1(x)
  ln_bf16_kernel<<<MROWS, 256, 0, stream>>>(x, ln1_g, ln1_b, hbf);
  // qkv = h1 @ Wqkv  (scatter to q/k/v^T)
  gemm_qkv_kernel<<<dim3(MROWS / 128, 2304 / 64), 256, 0, stream>>>(hbf, wqkv_t, qb, kb, vtb);
  // attention
  attn_kernel<<<dim3(NSEQ / 16, BB * HEADS), 32, 0, stream>>>(qb, kb, vtb, attnb);
  // x2 = x + attn @ Wout + b_out
  gemm_std_kernel<<<dim3(MROWS / 128, DIMV / 64), 256, 0, stream>>>(
      attnb, DIMV, wout_t, DIMV, DIMV, DIMV, b_out, x, 0, x2, nullptr);
  // h2 = LN2(x2)
  ln_bf16_kernel<<<MROWS, 256, 0, stream>>>(x2, ln2_g, ln2_b, hbf);
  // ff1 = gelu(h2 @ Wff1 + b_ff1)
  gemm_std_kernel<<<dim3(MROWS / 128, 3072 / 64), 256, 0, stream>>>(
      hbf, DIMV, wff1_t, DIMV, 3072, DIMV, b_ff1, nullptr, 1, nullptr, ff1b);
  // out = x2 + ff1 @ Wff2 + b_ff2
  gemm_std_kernel<<<dim3(MROWS / 128, DIMV / 64), 256, 0, stream>>>(
      ff1b, 3072, wff2_t, 3072, DIMV, 3072, b_ff2, x2, 0, (float*)d_out, nullptr);
}
